// NodeEdgeProcessor_39642548142789
// MI455X (gfx1250) — matrix operand order repaired and checked by reading, compiled, not run on hardware
//
#include <hip/hip_runtime.h>
#include <hip/hip_bf16.h>

// ---------------------------------------------------------------------------
// CDNA5 (gfx1250) GNN message-passing step using v_wmma_f32_16x16x32_bf16.
// Wave32, 256-thread blocks (8 waves), each wave owns a 16x128 output tile.
// ---------------------------------------------------------------------------

#define LDIM 128       // feature width
#define TILE_M 128     // rows per block
#define KCH 32         // K-chunk per WMMA step
#define NTT 8          // 128/16 n-tiles per wave
#define NTHREADS 256

typedef __bf16 bf16_t;
typedef __attribute__((ext_vector_type(16))) __bf16 bfrag16;  // WMMA A/B fragment
typedef __attribute__((ext_vector_type(8)))  __bf16 bfx8;
typedef __attribute__((ext_vector_type(4)))  __bf16 bfx4;
typedef __attribute__((ext_vector_type(8)))  float  v8f;      // WMMA C/D fragment
typedef __attribute__((ext_vector_type(4)))  int    v4i;

union FragU { bfrag16 v; bfx8 h[2]; };

__device__ __forceinline__ v8f wmma_bf16(bfrag16 a, bfrag16 b, v8f c) {
  // 8 args: (neg_a, A, neg_b, B, c_mod, C, reuse_a, reuse_b)
  return __builtin_amdgcn_wmma_f32_16x16x32_bf16(false, a, false, b, (short)0, c,
                                                 false, false);
}

// gfx1250 async global->LDS copy path (ASYNCcnt-tracked), if the toolchain
// declares the builtins; otherwise fall back to sync load+ds_store.
// Builtin signature (from hipcc diagnostic): (v4i AS1*, v4i AS3*, Ii, Ii).
#if __has_builtin(__builtin_amdgcn_global_load_async_to_lds_b128) && \
    __has_builtin(__builtin_amdgcn_s_wait_asynccnt)
#define USE_ASYNC_LDS 1
__device__ __forceinline__ void async_copy16(const void* g, void* l) {
  __builtin_amdgcn_global_load_async_to_lds_b128(
      (__attribute__((address_space(1))) v4i*)g,
      (__attribute__((address_space(3))) v4i*)l, 0, 0);
}
__device__ __forceinline__ void async_wait0() {
  __builtin_amdgcn_s_wait_asynccnt(0);
}
#else
#define USE_ASYNC_LDS 0
__device__ __forceinline__ void async_wait0() {}
#endif

// Copy one 16-byte weight segment into LDS (async if available).
__device__ __forceinline__ void stage_w16(const bf16_t* wp, bf16_t* lp) {
#if USE_ASYNC_LDS
  async_copy16(wp, lp);
#else
  *(bfx8*)lp = *(const bfx8*)wp;
#endif
}

// 8 n-tile WMMA sweep with 1-tile-ahead B-fragment pipelining.
__device__ __forceinline__ void wmma_sweep(const bf16_t (*sB)[KCH], int l15,
                                           int khalf, const FragU& a,
                                           v8f acc[NTT]) {
  FragU b0, b1;
  b0.h[0] = *(const bfx8*)&sB[l15][khalf * 8];
  b0.h[1] = *(const bfx8*)&sB[l15][16 + khalf * 8];
#pragma unroll
  for (int t = 0; t < NTT; ++t) {
    if (t + 1 < NTT) {
      const int n = ((t + 1) << 4) + l15;
      b1.h[0] = *(const bfx8*)&sB[n][khalf * 8];
      b1.h[1] = *(const bfx8*)&sB[n][16 + khalf * 8];
    }
    acc[t] = wmma_bf16(a.v, b0.v, acc[t]);
    b0 = b1;
  }
}

// ---------------------------------------------------------------------------
// Weight convert + transpose: fp32 [K][128] row-major -> bf16 [128][K]
// ---------------------------------------------------------------------------
__global__ void convert_wT_kernel(const float* __restrict__ w,
                                  bf16_t* __restrict__ out, int K) {
  const int n = blockIdx.x;  // 0..127 output column -> row of transposed
  for (int k = threadIdx.x; k < K; k += blockDim.x)
    out[(long long)n * K + k] = (bf16_t)w[(long long)k * LDIM + n];
}

__global__ void zero_kernel(float* __restrict__ p, long long n) {
  const long long i = (long long)blockIdx.x * blockDim.x + threadIdx.x;
  if (i < n) p[i] = 0.f;
}

// Fold BN batch stats + gamma/beta into per-column affine (bias pre-BN cancels).
__global__ void bn_finalize_kernel(const float* __restrict__ stats,
                                   const float* __restrict__ g,
                                   const float* __restrict__ bt,
                                   float inv_cnt, float* __restrict__ affine) {
  const int n = threadIdx.x;  // 128 threads
  const float mu  = stats[n] * inv_cnt;
  const float var = stats[LDIM + n] * inv_cnt - mu * mu;
  const float a   = g[n] * rsqrtf(var + 1e-5f);
  affine[n]        = a;
  affine[LDIM + n] = bt[n] - mu * a;
}

// ---------------------------------------------------------------------------
// Edge GEMM1: h1[e,:] = concat(x[tgt], x[src], edge_attr)[e,:] @ w1  (pre-bias)
// + per-column sum / sumsq for BN batch stats.
// ---------------------------------------------------------------------------
__global__ __launch_bounds__(NTHREADS)
void edge_lin1_kernel(const float* __restrict__ x,
                      const float* __restrict__ eattr,
                      const long long* __restrict__ eidx,
                      long long E,
                      const bf16_t* __restrict__ w1T,   // [128][384] bf16
                      float* __restrict__ h1,           // [E][128]
                      float* __restrict__ stats)        // [256] sum|sumsq
{
  __shared__ bf16_t sA[TILE_M][KCH];
  __shared__ bf16_t sB[LDIM][KCH];
  __shared__ int sSrc[TILE_M], sTgt[TILE_M];
  __shared__ float sSum[LDIM], sSq[LDIM];

  const int tid = threadIdx.x;
  const long long ebase = (long long)blockIdx.x * TILE_M;

  if (tid < TILE_M) {
    long long e = ebase + tid; if (e >= E) e = E - 1;
    sSrc[tid] = (int)eidx[e];
    sTgt[tid] = (int)eidx[E + e];
  }
  if (tid < LDIM) { sSum[tid] = 0.f; sSq[tid] = 0.f; }

  const int wave  = tid >> 5;
  const int lane  = tid & 31;
  const int l15   = lane & 15;
  const int khalf = lane >> 4;
  const int mrow  = (wave << 4) + l15;

  v8f acc[NTT] = {};
  __syncthreads();

  const int KDIM = 3 * LDIM;
  for (int kc = 0; kc < KDIM; kc += KCH) {
    const int piece   = kc >> 7;          // 0: x[tgt], 1: x[src], 2: edge_attr
    const int colbase = kc & (LDIM - 1);
#pragma unroll
    for (int it = 0; it < 4; ++it) {      // stage A: 128x32 fp32 -> bf16 LDS
      const int idx = it * NTHREADS + tid;
      const int row = idx >> 3;
      const int c4  = (idx & 7) << 2;
      bfx4 b4 = {(bf16_t)0.f, (bf16_t)0.f, (bf16_t)0.f, (bf16_t)0.f};
      if (ebase + row < E) {
        const float* rp;
        if (piece == 0)      rp = x + (long long)sTgt[row] * LDIM;
        else if (piece == 1) rp = x + (long long)sSrc[row] * LDIM;
        else                 rp = eattr + (ebase + row) * LDIM;
        const float4 v = *(const float4*)(rp + colbase + c4);
        b4 = bfx4{(bf16_t)v.x, (bf16_t)v.y, (bf16_t)v.z, (bf16_t)v.w};
      }
      *(bfx4*)&sA[row][c4] = b4;
    }
#pragma unroll
    for (int it = 0; it < 2; ++it) {      // stage B: transposed weight chunk
      const int idx = it * NTHREADS + tid;
      const int n = idx >> 2, seg = (idx & 3) << 3;
      const bf16_t* wp = w1T + (long long)n * KDIM + kc + seg;
      stage_w16(wp, &sB[n][seg]);
      __builtin_prefetch(wp + KCH, 0, 3);  // global_prefetch_b8: next K-chunk
    }
    async_wait0();
    __syncthreads();

    FragU a;
    a.h[0] = *(const bfx8*)&sA[mrow][khalf * 8];
    a.h[1] = *(const bfx8*)&sA[mrow][16 + khalf * 8];
    wmma_sweep(sB, l15, khalf, a, acc);
    __syncthreads();
  }

  // epilogue: store pre-bias activations + reduce column stats
#pragma unroll
  for (int t = 0; t < NTT; ++t) {
    const int n = (t << 4) + l15;
    float s = 0.f, q = 0.f;
#pragma unroll
    for (int r = 0; r < 8; ++r) {
      const int m = (wave << 4) + (khalf << 3) + r;
      const float v = acc[t][r];
      if (ebase + m < E) h1[(ebase + m) * LDIM + n] = v;
      s += v; q += v * v;
    }
    s += __shfl_xor(s, 16);
    q += __shfl_xor(q, 16);
    if (khalf == 0) { atomicAdd(&sSum[n], s); atomicAdd(&sSq[n], q); }
  }
  __syncthreads();
  if (tid < LDIM) {
    atomicAdd(&stats[tid], sSum[tid]);
    atomicAdd(&stats[LDIM + tid], sSq[tid]);
  }
}

// ---------------------------------------------------------------------------
// Node GEMM1: h1[i,:] = concat(x, agg)[i,:] @ nw1  (pre-bias) + stats
// ---------------------------------------------------------------------------
__global__ __launch_bounds__(NTHREADS)
void node_lin1_kernel(const float* __restrict__ x,
                      const float* __restrict__ agg,
                      long long N,
                      const bf16_t* __restrict__ w1T,   // [128][256]
                      float* __restrict__ h1,
                      float* __restrict__ stats)
{
  __shared__ bf16_t sA[TILE_M][KCH];
  __shared__ bf16_t sB[LDIM][KCH];
  __shared__ float sSum[LDIM], sSq[LDIM];

  const int tid = threadIdx.x;
  const long long nbase = (long long)blockIdx.x * TILE_M;
  if (tid < LDIM) { sSum[tid] = 0.f; sSq[tid] = 0.f; }

  const int wave  = tid >> 5;
  const int lane  = tid & 31;
  const int l15   = lane & 15;
  const int khalf = lane >> 4;
  const int mrow  = (wave << 4) + l15;

  v8f acc[NTT] = {};
  __syncthreads();

  const int KDIM = 2 * LDIM;
  for (int kc = 0; kc < KDIM; kc += KCH) {
    const int piece   = kc >> 7;          // 0: x, 1: agg
    const int colbase = kc & (LDIM - 1);
#pragma unroll
    for (int it = 0; it < 4; ++it) {
      const int idx = it * NTHREADS + tid;
      const int row = idx >> 3;
      const int c4  = (idx & 7) << 2;
      bfx4 b4 = {(bf16_t)0.f, (bf16_t)0.f, (bf16_t)0.f, (bf16_t)0.f};
      const long long g = nbase + row;
      if (g < N) {
        const float* rp = (piece == 0 ? x : agg) + g * LDIM;
        const float4 v = *(const float4*)(rp + colbase + c4);
        b4 = bfx4{(bf16_t)v.x, (bf16_t)v.y, (bf16_t)v.z, (bf16_t)v.w};
      }
      *(bfx4*)&sA[row][c4] = b4;
    }
#pragma unroll
    for (int it = 0; it < 2; ++it) {
      const int idx = it * NTHREADS + tid;
      const int n = idx >> 2, seg = (idx & 3) << 3;
      stage_w16(w1T + (long long)n * KDIM + kc + seg, &sB[n][seg]);
    }
    async_wait0();
    __syncthreads();

    FragU a;
    a.h[0] = *(const bfx8*)&sA[mrow][khalf * 8];
    a.h[1] = *(const bfx8*)&sA[mrow][16 + khalf * 8];
    wmma_sweep(sB, l15, khalf, a, acc);
    __syncthreads();
  }

#pragma unroll
  for (int t = 0; t < NTT; ++t) {
    const int n = (t << 4) + l15;
    float s = 0.f, q = 0.f;
#pragma unroll
    for (int r = 0; r < 8; ++r) {
      const int m = (wave << 4) + (khalf << 3) + r;
      const float v = acc[t][r];
      if (nbase + m < N) h1[(nbase + m) * LDIM + n] = v;
      s += v; q += v * v;   // padded rows contributed zero A -> zero acc
    }
    s += __shfl_xor(s, 16);
    q += __shfl_xor(q, 16);
    if (khalf == 0) { atomicAdd(&sSum[n], s); atomicAdd(&sSq[n], q); }
  }
  __syncthreads();
  if (tid < LDIM) {
    atomicAdd(&stats[tid], sSum[tid]);
    atomicAdd(&stats[LDIM + tid], sSq[tid]);
  }
}

// ---------------------------------------------------------------------------
// GEMM2: out = relu( relu(affine(h1)) @ w2 + b2 ) + resid
// EDGE: also atomicAdd(agg[tgt[e]], out)   (scatter-sum, L2 atomics)
// ---------------------------------------------------------------------------
template <bool EDGE>
__global__ __launch_bounds__(NTHREADS)
void lin2_kernel(const float* __restrict__ h1,
                 const float* __restrict__ affine,   // a[0..127] | b[128..255]
                 const bf16_t* __restrict__ w2T,     // [128][128]
                 const float* __restrict__ b2,
                 const float* __restrict__ resid,
                 float* __restrict__ out,
                 float* __restrict__ agg,
                 const long long* __restrict__ eidx,
                 long long E, long long nrows)
{
  __shared__ bf16_t sA[TILE_M][KCH];
  __shared__ bf16_t sB[LDIM][KCH];
  __shared__ int sTgt[TILE_M];

  const int tid = threadIdx.x;
  const long long base = (long long)blockIdx.x * TILE_M;
  if (EDGE && tid < TILE_M) {
    long long e = base + tid; if (e >= E) e = E - 1;
    sTgt[tid] = (int)eidx[E + e];
  }

  const int wave  = tid >> 5;
  const int lane  = tid & 31;
  const int l15   = lane & 15;
  const int khalf = lane >> 4;
  const int mrow  = (wave << 4) + l15;

  v8f acc[NTT] = {};
  __syncthreads();

  for (int kc = 0; kc < LDIM; kc += KCH) {
#pragma unroll
    for (int it = 0; it < 4; ++it) {      // stage A = relu(a*h1+b) in bf16
      const int idx = it * NTHREADS + tid;
      const int row = idx >> 3;
      const int c4  = (idx & 7) << 2;
      const long long g = base + row;
      float4 v = make_float4(0.f, 0.f, 0.f, 0.f);
      if (g < nrows) v = *(const float4*)(h1 + g * LDIM + kc + c4);
      const int n0 = kc + c4;
      bfx4 b4 = {
        (bf16_t)fmaxf(v.x * affine[n0 + 0] + affine[LDIM + n0 + 0], 0.f),
        (bf16_t)fmaxf(v.y * affine[n0 + 1] + affine[LDIM + n0 + 1], 0.f),
        (bf16_t)fmaxf(v.z * affine[n0 + 2] + affine[LDIM + n0 + 2], 0.f),
        (bf16_t)fmaxf(v.w * affine[n0 + 3] + affine[LDIM + n0 + 3], 0.f)};
      *(bfx4*)&sA[row][c4] = b4;
    }
#pragma unroll
    for (int it = 0; it < 2; ++it) {
      const int idx = it * NTHREADS + tid;
      const int n = idx >> 2, seg = (idx & 3) << 3;
      stage_w16(w2T + (long long)n * LDIM + kc + seg, &sB[n][seg]);
    }
    async_wait0();
    __syncthreads();

    FragU a;
    a.h[0] = *(const bfx8*)&sA[mrow][khalf * 8];
    a.h[1] = *(const bfx8*)&sA[mrow][16 + khalf * 8];
    wmma_sweep(sB, l15, khalf, a, acc);
    __syncthreads();
  }

#pragma unroll
  for (int t = 0; t < NTT; ++t) {
    const int n = (t << 4) + l15;
    const float bias = b2[n];
#pragma unroll
    for (int r = 0; r < 8; ++r) {
      const int m = (wave << 4) + (khalf << 3) + r;
      const long long g = base + m;
      if (g < nrows) {
        const float v = fmaxf(acc[t][r] + bias, 0.f) + resid[g * LDIM + n];
        out[g * LDIM + n] = v;
        if (EDGE) atomicAdd(agg + (long long)sTgt[m] * LDIM + n, v);
      }
    }
  }
}

// ---------------------------------------------------------------------------
// Host orchestration
// ---------------------------------------------------------------------------
extern "C" void kernel_launch(void* const* d_in, const int* in_sizes, int n_in,
                              void* d_out, int out_size, void* d_ws, size_t ws_size,
                              hipStream_t stream) {
  const long long N = in_sizes[0] / LDIM;
  const long long E = in_sizes[1] / 2;
  const int STEPS = in_sizes[3] / (3 * LDIM * LDIM);

  const float* x0in      = (const float*)d_in[0];
  const long long* eidx  = (const long long*)d_in[1];
  const float* ea0       = (const float*)d_in[2];
  const float* ew1 = (const float*)d_in[3];
  const float* eg1 = (const float*)d_in[5];
  const float* ebt1= (const float*)d_in[6];
  const float* ew2 = (const float*)d_in[7];
  const float* eb2 = (const float*)d_in[8];
  const float* nw1 = (const float*)d_in[9];
  const float* ng1 = (const float*)d_in[11];
  const float* nbt1= (const float*)d_in[12];
  const float* nw2 = (const float*)d_in[13];
  const float* nb2 = (const float*)d_in[14];

  // workspace carve
  uintptr_t p = (uintptr_t)d_ws;
  auto carve = [&](size_t bytes) -> void* {
    void* r = (void*)p; p += (bytes + 255) & ~(size_t)255; return r;
  };
  const size_t PS = (size_t)LDIM * (3 * LDIM) + (size_t)LDIM * LDIM +
                    (size_t)LDIM * (2 * LDIM) + (size_t)LDIM * LDIM;  // bf16/step
  bf16_t* wbf  = (bf16_t*)carve((size_t)STEPS * PS * sizeof(bf16_t));
  float* h1    = (float*)carve((size_t)E * LDIM * sizeof(float));  // reused by node
  float* ue    = (float*)carve((size_t)E * LDIM * sizeof(float));
  float* aggb  = (float*)carve((size_t)N * LDIM * sizeof(float));
  float* xbuf  = (float*)carve((size_t)N * LDIM * sizeof(float));
  float* stats = (float*)carve(2 * LDIM * sizeof(float));
  float* affn  = (float*)carve(2 * LDIM * sizeof(float));

  // weights -> transposed bf16 (once per call; tiny)
  for (int s = 0; s < STEPS; ++s) {
    bf16_t* b    = wbf + (size_t)s * PS;
    bf16_t* ew1T = b;
    bf16_t* ew2T = ew1T + (size_t)LDIM * 3 * LDIM;
    bf16_t* nw1T = ew2T + (size_t)LDIM * LDIM;
    bf16_t* nw2T = nw1T + (size_t)LDIM * 2 * LDIM;
    convert_wT_kernel<<<LDIM, 128, 0, stream>>>(ew1 + (size_t)s*3*LDIM*LDIM, ew1T, 3*LDIM);
    convert_wT_kernel<<<LDIM, 128, 0, stream>>>(ew2 + (size_t)s*LDIM*LDIM,   ew2T, LDIM);
    convert_wT_kernel<<<LDIM, 128, 0, stream>>>(nw1 + (size_t)s*2*LDIM*LDIM, nw1T, 2*LDIM);
    convert_wT_kernel<<<LDIM, 128, 0, stream>>>(nw2 + (size_t)s*LDIM*LDIM,   nw2T, LDIM);
  }

  float* out_x  = (float*)d_out;
  float* out_ea = (float*)d_out + ((size_t)out_size - (size_t)E * LDIM);

  const int eblocks = (int)((E + TILE_M - 1) / TILE_M);
  const int nblocks = (int)((N + TILE_M - 1) / TILE_M);
  const long long aggN = N * LDIM;

  const float* cur_x  = x0in;
  const float* cur_ea = ea0;
  for (int s = 0; s < STEPS; ++s) {
    bf16_t* b    = wbf + (size_t)s * PS;
    bf16_t* ew1T = b;
    bf16_t* ew2T = ew1T + (size_t)LDIM * 3 * LDIM;
    bf16_t* nw1T = ew2T + (size_t)LDIM * LDIM;
    bf16_t* nw2T = nw1T + (size_t)LDIM * 2 * LDIM;
    float* ueo = (s == STEPS - 1) ? out_ea : ue;
    float* xo  = (s == STEPS - 1) ? out_x  : xbuf;

    zero_kernel<<<1, 256, 0, stream>>>(stats, 2 * LDIM);
    zero_kernel<<<(int)((aggN + 255) / 256), 256, 0, stream>>>(aggb, aggN);

    edge_lin1_kernel<<<eblocks, NTHREADS, 0, stream>>>(
        cur_x, cur_ea, eidx, E, ew1T, h1, stats);
    bn_finalize_kernel<<<1, LDIM, 0, stream>>>(
        stats, eg1 + (size_t)s*LDIM, ebt1 + (size_t)s*LDIM, 1.f/(float)E, affn);
    lin2_kernel<true><<<eblocks, NTHREADS, 0, stream>>>(
        h1, affn, ew2T, eb2 + (size_t)s*LDIM, cur_ea, ueo, aggb, eidx, E, E);

    zero_kernel<<<1, 256, 0, stream>>>(stats, 2 * LDIM);
    node_lin1_kernel<<<nblocks, NTHREADS, 0, stream>>>(
        cur_x, aggb, N, nw1T, h1, stats);
    bn_finalize_kernel<<<1, LDIM, 0, stream>>>(
        stats, ng1 + (size_t)s*LDIM, nbt1 + (size_t)s*LDIM, 1.f/(float)N, affn);
    lin2_kernel<false><<<nblocks, NTHREADS, 0, stream>>>(
        h1, affn, nw2T, nb2 + (size_t)s*LDIM, cur_x, xo, nullptr, nullptr, 0, N);

    cur_x = xo; cur_ea = ueo;
  }

  // pass-through edge_index into the middle of d_out (raw bytes)
  size_t midElems = (size_t)out_size - (size_t)N * LDIM - (size_t)E * LDIM;
  size_t copyBytes = midElems * sizeof(float);
  size_t srcBytes = (size_t)2 * E * sizeof(long long);
  if (copyBytes > srcBytes) copyBytes = srcBytes;
  (void)hipMemcpyAsync((char*)d_out + (size_t)N * LDIM * sizeof(float), d_in[1],
                       copyBytes, hipMemcpyDeviceToDevice, stream);
}
